// GRUModel_13529146982599
// MI455X (gfx1250) — compile-verified, matrix operand order checked by reading, not run on hardware
//
#include <hip/hip_runtime.h>

typedef __bf16 bf16;
typedef __attribute__((ext_vector_type(16))) bf16  v16bf;
typedef __attribute__((ext_vector_type(8)))  float v8f;
typedef __attribute__((ext_vector_type(4)))  unsigned int v4u;
typedef __attribute__((ext_vector_type(4)))  float v4f;

#define B_ 4096
#define T_ 128
#define D_ 32
#define H_ 64
#define G_ 192   // 3*H
#define WAVES 2

__device__ __forceinline__ v8f wmma_bf16(v16bf a, v16bf b, v8f c) {
  return __builtin_amdgcn_wmma_f32_16x16x32_bf16(false, a, false, b, (short)0, c,
                                                 false, false);
}

// Load a 16-half A/B fragment from a row-major bf16 LDS tile.
// Lane (hh=lane>>4, idx=lane&15): VGPR0-3 = K k0+hh*8 .. +7, VGPR4-7 = K k0+16+hh*8 .. +7
__device__ __forceinline__ v16bf load_frag(const bf16* base, int ldk, int r, int k0, int hh) {
  const bf16* p = base + r * ldk + k0 + hh * 8;
  union { v4u u[2]; v16bf b; } u;
  u.u[0] = *(const v4u*)(p);
  u.u[1] = *(const v4u*)(p + 16);
  return u.b;
}

__device__ __forceinline__ v8f splat8(float v) { return (v8f){v, v, v, v, v, v, v, v}; }

// Fast reciprocal: single v_rcp_f32 instead of IEEE-division Newton sequence.
__device__ __forceinline__ float fast_rcp(float x) {
#if __has_builtin(__builtin_amdgcn_rcpf)
  return __builtin_amdgcn_rcpf(x);
#else
  return 1.0f / x;
#endif
}

// sigmoid(x) = rcp(1 + exp(-x)); exp overflow -> inf -> rcp -> 0 (correct limit).
__device__ __forceinline__ float sigm(float x) {
  return fast_rcp(1.0f + __expf(-x));
}

__device__ __forceinline__ float tanh_f(float x) {
#if __has_builtin(__builtin_amdgcn_tanhf)
  return __builtin_amdgcn_tanhf(x);      // gfx1250 v_tanh_f32 (TRANS op)
#else
  x = fminf(fmaxf(x, -15.0f), 15.0f);
  float e = __expf(-2.0f * x);
  return (1.0f - e) * fast_rcp(1.0f + e);
#endif
}

// One GRU step for a 16-row batch tile. Gates laid out [r | z | n], 12 N-tiles of 16.
// hc[4] = f32 hidden state in WMMA C layout. New h is written back to sH (bf16, 16x64).
template<int DIN>
__device__ __forceinline__ void gru_step(
    const v16bf* xf,                 // DIN/32 A-fragments of the layer input
    bf16* sH,                        // this layer's hidden-state stage (wave-private)
    const bf16* sWih, const bf16* sWhh,
    const float* brz, const float* bxn, const float* bhn,  // per-lane bias scalars
    v8f* hc, int row, int hh, int col)
{
  v16bf h0 = load_frag(sH, H_, row, 0, hh);
  v16bf h1 = load_frag(sH, H_, row, 32, hh);
#pragma unroll
  for (int jt = 0; jt < 4; ++jt) {
    v8f ar = splat8(brz[jt]);        // r-gate acc (x + h contributions share it)
    v8f az = splat8(brz[4 + jt]);    // z-gate acc
    v8f an = splat8(bxn[jt]);        // x-part of n-gate
    v8f ah = splat8(bhn[jt]);        // h-part of n-gate (kept separate: needs r*)
#pragma unroll
    for (int kf = 0; kf < DIN / 32; ++kf) {
      ar = wmma_bf16(xf[kf], load_frag(sWih, DIN, (0 + jt) * 16 + col, kf * 32, hh), ar);
      az = wmma_bf16(xf[kf], load_frag(sWih, DIN, (4 + jt) * 16 + col, kf * 32, hh), az);
      an = wmma_bf16(xf[kf], load_frag(sWih, DIN, (8 + jt) * 16 + col, kf * 32, hh), an);
    }
    ar = wmma_bf16(h0, load_frag(sWhh, H_, (0 + jt) * 16 + col, 0, hh), ar);
    ar = wmma_bf16(h1, load_frag(sWhh, H_, (0 + jt) * 16 + col, 32, hh), ar);
    az = wmma_bf16(h0, load_frag(sWhh, H_, (4 + jt) * 16 + col, 0, hh), az);
    az = wmma_bf16(h1, load_frag(sWhh, H_, (4 + jt) * 16 + col, 32, hh), az);
    ah = wmma_bf16(h0, load_frag(sWhh, H_, (8 + jt) * 16 + col, 0, hh), ah);
    ah = wmma_bf16(h1, load_frag(sWhh, H_, (8 + jt) * 16 + col, 32, hh), ah);
#pragma unroll
    for (int v = 0; v < 8; ++v) {
      float r = sigm(ar[v]);
      float z = sigm(az[v]);
      float n = tanh_f(an[v] + r * ah[v]);
      float h = n + z * (hc[jt][v] - n);   // (1-z)*n + z*h, 2 ops
      hc[jt][v] = h;
      // C layout: row = v + 8*hh, col = jt*16 + (lane&15)
      sH[(v + 8 * hh) * H_ + jt * 16 + col] = (bf16)h;
    }
  }
}

__global__ __launch_bounds__(WAVES * 32) void gru_fused_kernel(
    const float* __restrict__ x,
    const float* __restrict__ Wih0, const float* __restrict__ Whh0,
    const float* __restrict__ bih0, const float* __restrict__ bhh0,
    const float* __restrict__ Wih1, const float* __restrict__ Whh1,
    const float* __restrict__ bih1, const float* __restrict__ bhh1,
    float* __restrict__ hlast)
{
  __shared__ __align__(16) bf16 sWih0[G_ * D_];
  __shared__ __align__(16) bf16 sWhh0[G_ * H_];
  __shared__ __align__(16) bf16 sWih1[G_ * H_];
  __shared__ __align__(16) bf16 sWhh1[G_ * H_];
  __shared__ __align__(16) bf16 sHA[WAVES][16 * H_];
  __shared__ __align__(16) bf16 sHB[WAVES][16 * H_];

  const int tid  = threadIdx.x;
  const int w    = tid >> 5;
  const int lane = tid & 31;
  const int hh   = lane >> 4;
  const int row  = lane & 15;
  const int col  = lane & 15;

  // Stage weights in LDS as bf16 (shared by both waves in the block).
  for (int i = tid; i < G_ * D_; i += WAVES * 32) sWih0[i] = (bf16)Wih0[i];
  for (int i = tid; i < G_ * H_; i += WAVES * 32) sWhh0[i] = (bf16)Whh0[i];
  for (int i = tid; i < G_ * H_; i += WAVES * 32) sWih1[i] = (bf16)Wih1[i];
  for (int i = tid; i < G_ * H_; i += WAVES * 32) sWhh1[i] = (bf16)Whh1[i];
  for (int i = lane; i < 16 * H_; i += 32) {
    sHA[w][i] = (bf16)0.0f;
    sHB[w][i] = (bf16)0.0f;
  }
  __syncthreads();

  // Per-lane bias scalars: column = lane&15 of each gate N-tile.
  float brz0[8], bxn0[4], bhn0[4], brz1[8], bxn1[4], bhn1[4];
#pragma unroll
  for (int nt = 0; nt < 8; ++nt) {
    brz0[nt] = bih0[nt * 16 + col] + bhh0[nt * 16 + col];
    brz1[nt] = bih1[nt * 16 + col] + bhh1[nt * 16 + col];
  }
#pragma unroll
  for (int jt = 0; jt < 4; ++jt) {
    bxn0[jt] = bih0[128 + jt * 16 + col];
    bhn0[jt] = bhh0[128 + jt * 16 + col];
    bxn1[jt] = bih1[128 + jt * 16 + col];
    bhn1[jt] = bhh1[128 + jt * 16 + col];
  }

  const int b0 = (blockIdx.x * WAVES + w) * 16;
  v8f hA[4], hB[4];
#pragma unroll
  for (int jt = 0; jt < 4; ++jt) { hA[jt] = splat8(0.0f); hB[jt] = splat8(0.0f); }

  const float* xr = x + (long)(b0 + row) * T_ * D_;

#pragma unroll 1
  for (int t = 0; t < T_; ++t) {
    const float* xp = xr + (long)t * D_;
    if (t + 1 < T_) __builtin_prefetch(xp + D_, 0, 1);   // global_prefetch next step

    // Build layer-0 x A-fragment (16x32 f32 -> bf16) straight from global.
    union { v4f q[4]; float f[16]; } u;
    u.q[0] = *(const v4f*)(xp + hh * 8);
    u.q[1] = *(const v4f*)(xp + hh * 8 + 4);
    u.q[2] = *(const v4f*)(xp + 16 + hh * 8);
    u.q[3] = *(const v4f*)(xp + 16 + hh * 8 + 4);
    v16bf xf0;
#pragma unroll
    for (int i = 0; i < 16; ++i) xf0[i] = (bf16)u.f[i];
    v16bf xfa[1] = { xf0 };

    // Layer 0 advances; its fresh h lands in sHA (wave-private, same-wave DS order).
    gru_step<D_>(xfa, sHA[w], sWih0, sWhh0, brz0, bxn0, bhn0, hA, row, hh, col);

    // Layer 1 consumes layer 0's new hidden state immediately (no HBM intermediate).
    v16bf xf1[2];
    xf1[0] = load_frag(sHA[w], H_, row, 0, hh);
    xf1[1] = load_frag(sHA[w], H_, row, 32, hh);

    gru_step<H_>(xf1, sHB[w], sWih1, sWhh1, brz1, bxn1, bhn1, hB, row, hh, col);
  }

  // Spill final layer-1 hidden state (f32) for the FC kernel.
#pragma unroll
  for (int jt = 0; jt < 4; ++jt) {
#pragma unroll
    for (int v = 0; v < 8; ++v) {
      hlast[(long)(b0 + v + 8 * hh) * H_ + jt * 16 + col] = hB[jt][v];
    }
  }
}

__global__ void fc_kernel(const float* __restrict__ hlast,
                          const float* __restrict__ fcw,
                          const float* __restrict__ fcb,
                          float* __restrict__ out)
{
  int i = blockIdx.x * blockDim.x + threadIdx.x;
  if (i >= B_) return;
  float s = fcb[0];
#pragma unroll
  for (int j = 0; j < H_; ++j) s += hlast[(long)i * H_ + j] * fcw[j];
  out[i] = s;
}

extern "C" void kernel_launch(void* const* d_in, const int* in_sizes, int n_in,
                              void* d_out, int out_size, void* d_ws, size_t ws_size,
                              hipStream_t stream) {
  (void)in_sizes; (void)n_in; (void)out_size; (void)ws_size;
  const float* x    = (const float*)d_in[0];
  const float* Wih0 = (const float*)d_in[1];
  const float* Whh0 = (const float*)d_in[2];
  const float* bih0 = (const float*)d_in[3];
  const float* bhh0 = (const float*)d_in[4];
  const float* Wih1 = (const float*)d_in[5];
  const float* Whh1 = (const float*)d_in[6];
  const float* bih1 = (const float*)d_in[7];
  const float* bhh1 = (const float*)d_in[8];
  const float* fcw  = (const float*)d_in[9];
  const float* fcb  = (const float*)d_in[10];
  float* out   = (float*)d_out;
  float* hlast = (float*)d_ws;   // B_*H_ floats = 1 MB scratch

  dim3 grid(B_ / (WAVES * 16));  // 128 blocks x 2 waves x 16 rows = 4096
  gru_fused_kernel<<<grid, WAVES * 32, 0, stream>>>(
      x, Wih0, Whh0, bih0, bhh0, Wih1, Whh1, bih1, bhh1, hlast);

  fc_kernel<<<(B_ + 255) / 256, 256, 0, stream>>>(hlast, fcw, fcb, out);
}
